// PerturbedRank_73297911873658
// MI455X (gfx1250) — matrix operand order repaired and checked by reading, not compiled, not run on hardware
//
#include <hip/hip_runtime.h>
#include <hip/hip_bf16.h>
#include <math.h>

// ---------------- problem constants ----------------
constexpr int Bn = 32, Cn = 384, Tn = 16, Hn = 14, Wn = 14;
constexpr int HO = 6, WO = 6;                  // (14-3)/2+1
constexpr int KCONV = Cn * 9;                  // 3456  (GEMM K for conv)
constexpr int MCONV = Bn * Tn * HO * WO;       // 18432 (GEMM M for conv)
constexpr int C2 = 2 * Cn;                     // 768
constexpr int BT = Bn * Tn;                    // 512
constexpr int KSEL = 8;
constexpr int NSAMP = 1000;
constexpr float SIGMA = 0.05f;
constexpr int FRAME_OUT = Bn * Cn * KSEL * Hn * Wn;  // 19267584 per output tensor

// GEMM tiling: block = 256 threads = 8 waves; block tile 128(M) x 64(N), K-depth 64.
// Waves in 4x2 grid, each wave owns a 32x32 C tile = 2x2 WMMA sub-tiles.
// Double-buffered LDS; B tiles arrive via async global->LDS DMA (ASYNCcnt).
constexpr int BM = 128, BN = 64, BK = 64;

// ---------------- WMMA vector types ----------------
typedef __attribute__((ext_vector_type(16))) __bf16 v16bf;
typedef __attribute__((ext_vector_type(16))) unsigned short v16us;
typedef __attribute__((ext_vector_type(8)))  float v8f;
typedef int v4i_t __attribute__((vector_size(16)));   // matches async-LDS builtin param

#define AS1 __attribute__((address_space(1)))
#define AS3 __attribute__((address_space(3)))

#if defined(__gfx1250__) && __has_builtin(__builtin_amdgcn_global_load_async_to_lds_b128)
#define HAVE_ASYNC_LDS 1
#else
#define HAVE_ASYNC_LDS 0
#endif

__device__ __forceinline__ void wait_asynccnt0() {
#if __has_builtin(__builtin_amdgcn_s_wait_asynccnt)
  __builtin_amdgcn_s_wait_asynccnt(0);
#else
  asm volatile("s_wait_asynccnt 0" ::: "memory");
#endif
}

__device__ __forceinline__ unsigned short f2bf(float f) {
  union { float f; unsigned u; } v; v.f = f;
  unsigned u = v.u;
  unsigned r = u + 0x7FFFu + ((u >> 16) & 1u);   // round-to-nearest-even
  return (unsigned short)(r >> 16);
}

__device__ __forceinline__ float gelu_exact(float x) {
  return 0.5f * x * (1.0f + erff(x * 0.70710678118654752f));
}

// Issue async DMA (or sync copy) of a BK x BN bf16 tile of Bg ([Kd x N]) into LDS.
__device__ __forceinline__ void stage_b_issue(const unsigned short* __restrict__ Bg,
                                              unsigned short* Bs, int k0, int blockN,
                                              int N, int tid) {
#if HAVE_ASYNC_LDS
  const int r  = tid >> 3;          // 0..31, two rows per thread (r, r+32)
  const int c8 = (tid & 7) * 8;     // bf16 column offset (16B granules)
  const unsigned short* s0 = Bg + (size_t)(k0 + r) * N + blockN + c8;
  const unsigned short* s1 = Bg + (size_t)(k0 + r + 32) * N + blockN + c8;
  __builtin_amdgcn_global_load_async_to_lds_b128((AS1 v4i_t*)s0,
                                                 (AS3 v4i_t*)&Bs[r * BN + c8], 0, 0);
  __builtin_amdgcn_global_load_async_to_lds_b128((AS1 v4i_t*)s1,
                                                 (AS3 v4i_t*)&Bs[(r + 32) * BN + c8], 0, 0);
#else
  for (int vi = tid; vi < (BK * BN) / 8; vi += 256) {
    const int r = vi >> 3;
    const int c8 = (vi & 7) * 8;
    *(uint4*)&Bs[r * BN + c8] = *(const uint4*)&Bg[(size_t)(k0 + r) * N + blockN + c8];
  }
#endif
}

__device__ __forceinline__ void stage_b_wait() {
#if HAVE_ASYNC_LDS
  wait_asynccnt0();
#endif
}

// A fragment (16x32 bf16) from LDS. `As` points at row 0 / k-offset already applied.
// ISA 16-bit A layout: lanes 0-15 -> M=lane, K {0-7,16-23}; lanes 16-31 -> M=lane-16, K {8-15,24-31}.
__device__ __forceinline__ v16bf load_a_frag(const unsigned short* As, int lane) {
  const int half = lane >> 4, l15 = lane & 15;
  v16us au;
  const unsigned short* ap = As + l15 * BK;
#pragma unroll
  for (int j = 0; j < 8; ++j) {
    au[j]     = ap[half * 8 + j];
    au[8 + j] = ap[16 + half * 8 + j];
  }
  return __builtin_bit_cast(v16bf, au);
}

// B fragment (32x16 bf16) from LDS. `Bs` points at [ksub][n0].
// ISA B layout: lanes 0-15 hold K {0-7,16-23} of col N=lane; lanes 16-31 hold K {8-15,24-31}.
__device__ __forceinline__ v16bf load_b_frag(const unsigned short* Bs, int lane) {
  const int half = lane >> 4, l15 = lane & 15;
  v16us bu;
#pragma unroll
  for (int j = 0; j < 8; ++j) {
    bu[j]     = Bs[(half * 8 + j) * BN + l15];
    bu[8 + j] = Bs[(16 + half * 8 + j) * BN + l15];
  }
  return __builtin_bit_cast(v16bf, bu);
}

__device__ __forceinline__ v8f wmma_bf16(v16bf a, v16bf b, v8f c) {
  return __builtin_amdgcn_wmma_f32_16x16x32_bf16(false, a, false, b, (short)0, c,
                                                 false, false);
}

// Full K-64 tile of 2x2 WMMAs for one wave.
__device__ __forceinline__ void mma_tile(const unsigned short* As_,
                                         const unsigned short* Bs_,
                                         int lane, int wm, int wn, v8f acc[2][2]) {
#pragma unroll
  for (int ks = 0; ks < BK; ks += 32) {
    const v16bf a0 = load_a_frag(&As_[(wm * 32)      * BK + ks], lane);
    const v16bf a1 = load_a_frag(&As_[(wm * 32 + 16) * BK + ks], lane);
    const v16bf b0 = load_b_frag(&Bs_[ks * BN + wn * 32],      lane);
    const v16bf b1 = load_b_frag(&Bs_[ks * BN + wn * 32 + 16], lane);
    acc[0][0] = wmma_bf16(a0, b0, acc[0][0]);
    acc[0][1] = wmma_bf16(a0, b1, acc[0][1]);
    acc[1][0] = wmma_bf16(a1, b0, acc[1][0]);
    acc[1][1] = wmma_bf16(a1, b1, acc[1][1]);
  }
}

// ---------------- Kernel 0a: weights [K][N] fp32 -> bf16 (already row-major K x N) ----
__global__ void cvt_w_k(const float* __restrict__ src, unsigned short* __restrict__ dst,
                        int n) {
  const int i = blockIdx.x * 256 + threadIdx.x;
  if (i < n) dst[i] = f2bf(src[i]);
}

// ---------------- Kernel 0b: conv weights (Cout,Cin,1,3,3) -> bf16 [KCONV][Cn] --------
__global__ void cvt_conv_w_k(const float* __restrict__ src,
                             unsigned short* __restrict__ dst) {
  const int i = blockIdx.x * 256 + threadIdx.x;
  if (i < KCONV * Cn) {
    const int k = i / Cn, n = i - k * Cn;
    dst[(size_t)k * Cn + n] = f2bf(src[(size_t)n * KCONV + k]);
  }
}

// ---------------- Kernel 1: conv3d (1x3x3, stride 1,2,2) as implicit GEMM, WMMA bf16 ----
// out[m][n], m = ((b*T + t)*HO + h)*WO + w, n = c_out.  +conv bias.
// Double-buffered; address decode hoisted to per-block LDS tables.
__global__ __launch_bounds__(256)
void conv_wmma_k(const float* __restrict__ frames, const unsigned short* __restrict__ wg,
                 const float* __restrict__ conv_b, float* __restrict__ out) {
  __shared__ unsigned short As[2][BM * BK];   // 32 KB
  __shared__ unsigned short Bs[2][BK * BN];   // 16 KB
  __shared__ int mrow[BM];                    // frames offset for each tile row (k-invariant)
  __shared__ int ktab[KCONV];                 // frames offset for each k (m-invariant)
  const int tid  = threadIdx.x;
  const int lane = tid & 31;
  const int wave = tid >> 5;
  const int wm = wave & 3, wn = wave >> 2;
  const int blockM = blockIdx.y * BM;
  const int blockN = blockIdx.x * BN;

  // Build decode tables once per block.
  if (tid < BM) {
    const int gm = blockM + tid;
    const int b  = gm / (Tn * HO * WO);
    int rm = gm - b * (Tn * HO * WO);
    const int t  = rm / (HO * WO);
    rm -= t * (HO * WO);
    const int hh = rm / WO, ww = rm - (rm / WO) * WO;
    mrow[tid] = ((b * Cn) * Tn + t) * (Hn * Wn) + (2 * hh) * Wn + (2 * ww);
  }
  for (int gk = tid; gk < KCONV; gk += 256) {
    const int ci = gk / 9;
    const int r9 = gk - ci * 9;
    const int kh = r9 / 3, kw = r9 - (r9 / 3) * 3;
    ktab[gk] = ci * (Tn * Hn * Wn) + kh * Wn + kw;
  }
  __syncthreads();

  v8f acc[2][2];
#pragma unroll
  for (int i = 0; i < 2; ++i)
#pragma unroll
    for (int j = 0; j < 2; ++j) acc[i][j] = {0.f,0.f,0.f,0.f,0.f,0.f,0.f,0.f};

  const int cA = tid & 63, r0 = tid >> 6;   // fixed column / starting row for A staging
  // prologue: stage tile 0
  {
    const int kof = ktab[cA];
#pragma unroll 4
    for (int r = r0; r < BM; r += 4)
      As[0][r * BK + cA] = f2bf(frames[mrow[r] + kof]);
    stage_b_issue(wg, Bs[0], 0, blockN, Cn, tid);
    stage_b_wait();
  }
  __syncthreads();

  const int nk = KCONV / BK;
  for (int kt = 0; kt < nk; ++kt) {
    const int cur = kt & 1, nxt = cur ^ 1;
    if (kt + 1 < nk) {   // stage next tile while computing current
      const int k0n = (kt + 1) * BK;
      const int kof = ktab[k0n + cA];
#pragma unroll 4
      for (int r = r0; r < BM; r += 4)
        As[nxt][r * BK + cA] = f2bf(frames[mrow[r] + kof]);
      stage_b_issue(wg, Bs[nxt], k0n, blockN, Cn, tid);
    }
    mma_tile(As[cur], Bs[cur], lane, wm, wn, acc);
    if (kt + 1 < nk) stage_b_wait();
    __syncthreads();
  }

  const int half = lane >> 4, l15 = lane & 15;
#pragma unroll
  for (int im = 0; im < 2; ++im) {
#pragma unroll
    for (int in_ = 0; in_ < 2; ++in_) {
      const int col = blockN + wn * 32 + in_ * 16 + l15;
      const float bias = conv_b[col];
#pragma unroll
      for (int r = 0; r < 8; ++r) {
        const int row = blockM + wm * 32 + im * 16 + r + 8 * half;
        out[(size_t)row * Cn + col] = acc[im][in_][r] + bias;
      }
    }
  }
}

// ---------------- Kernel 2: generic fused GEMM  out = act(A @ W + bias), WMMA bf16 ----
// A: [M][Kd] fp32, Bg: [Kd][N] bf16 (pre-converted). Double-buffered.
__global__ __launch_bounds__(256)
void gemm_wmma_k(const float* __restrict__ A, const unsigned short* __restrict__ Bg,
                 const float* __restrict__ bias, float* __restrict__ out,
                 int M, int N, int Kd, int act) {
  __shared__ unsigned short As[2][BM * BK];
  __shared__ unsigned short Bs[2][BK * BN];
  const int tid  = threadIdx.x;
  const int lane = tid & 31;
  const int wave = tid >> 5;
  const int wm = wave & 3, wn = wave >> 2;
  const int blockM = blockIdx.y * BM;
  const int blockN = blockIdx.x * BN;
  v8f acc[2][2];
#pragma unroll
  for (int i = 0; i < 2; ++i)
#pragma unroll
    for (int j = 0; j < 2; ++j) acc[i][j] = {0.f,0.f,0.f,0.f,0.f,0.f,0.f,0.f};

  // prologue: stage tile 0
  for (int vi = tid; vi < (BM * BK) / 4; vi += 256) {
    const int r = vi >> 4;
    const int c4 = (vi & 15) * 4;
    const float4 v = *(const float4*)&A[(size_t)(blockM + r) * Kd + c4];
    unsigned short* dst = &As[0][r * BK + c4];
    dst[0] = f2bf(v.x); dst[1] = f2bf(v.y); dst[2] = f2bf(v.z); dst[3] = f2bf(v.w);
  }
  stage_b_issue(Bg, Bs[0], 0, blockN, N, tid);
  stage_b_wait();
  __syncthreads();

  const int nk = Kd / BK;
  for (int kt = 0; kt < nk; ++kt) {
    const int cur = kt & 1, nxt = cur ^ 1;
    if (kt + 1 < nk) {
      const int k0n = (kt + 1) * BK;
      if (kt + 2 < nk)  // hint the tile after next (global_prefetch_b8)
        __builtin_prefetch(A + (size_t)(blockM + (tid >> 1)) * Kd + k0n + BK +
                           (tid & 1) * 32, 0, 0);
      for (int vi = tid; vi < (BM * BK) / 4; vi += 256) {
        const int r = vi >> 4;
        const int c4 = (vi & 15) * 4;
        const float4 v = *(const float4*)&A[(size_t)(blockM + r) * Kd + k0n + c4];
        unsigned short* dst = &As[nxt][r * BK + c4];
        dst[0] = f2bf(v.x); dst[1] = f2bf(v.y); dst[2] = f2bf(v.z); dst[3] = f2bf(v.w);
      }
      stage_b_issue(Bg, Bs[nxt], k0n, blockN, N, tid);
    }
    mma_tile(As[cur], Bs[cur], lane, wm, wn, acc);
    if (kt + 1 < nk) stage_b_wait();
    __syncthreads();
  }

  const int half = lane >> 4, l15 = lane & 15;
#pragma unroll
  for (int im = 0; im < 2; ++im) {
#pragma unroll
    for (int in_ = 0; in_ < 2; ++in_) {
      const int col = blockN + wn * 32 + in_ * 16 + l15;
      const float bv = bias[col];
#pragma unroll
      for (int r = 0; r < 8; ++r) {
        const int row = blockM + wm * 32 + im * 16 + r + 8 * half;
        float v = acc[im][in_][r] + bv;
        if (act) v = gelu_exact(v);
        out[(size_t)row * N + col] = v;
      }
    }
  }
}

// ---------------- Kernel 3: per-channel BN partial sums (coalesced, atomic merge) -----
// grid (Cn/64, 32); consecutive lanes read consecutive channels.
__global__ void bn_stats_k(const float* __restrict__ x, float* __restrict__ sum,
                           float* __restrict__ sum2) {
  const int coff = threadIdx.x & 63, lm = threadIdx.x >> 6;   // 64 channels x 4 m-lanes
  const int c = blockIdx.x * 64 + coff;
  const int mchunk = MCONV / 32;
  const int m0 = blockIdx.y * mchunk;
  float s = 0.f, s2 = 0.f;
  for (int m = m0 + lm; m < m0 + mchunk; m += 4) {
    const float v = x[(size_t)m * Cn + c];
    s += v; s2 += v * v;
  }
  __shared__ float sh1[256], sh2[256];
  sh1[threadIdx.x] = s; sh2[threadIdx.x] = s2;
  __syncthreads();
  if (lm == 0) {
    s  = sh1[coff] + sh1[64 + coff] + sh1[128 + coff] + sh1[192 + coff];
    s2 = sh2[coff] + sh2[64 + coff] + sh2[128 + coff] + sh2[192 + coff];
    atomicAdd(&sum[c], s);
    atomicAdd(&sum2[c], s2);
  }
}

// ---------------- Kernel 4: BN-normalize + spatial mean/max pool -> (BT, 2C) ----------
__global__ void bnpool_k(const float* __restrict__ x, const float* __restrict__ sum,
                         const float* __restrict__ sum2, const float* __restrict__ g,
                         const float* __restrict__ be, float* __restrict__ pooled) {
  const int bt = blockIdx.x;  // 0..BT-1
  for (int c = threadIdx.x; c < Cn; c += blockDim.x) {
    const float mu = sum[c] * (1.f / MCONV);
    const float inv = rsqrtf(sum2[c] * (1.f / MCONV) - mu * mu + 1e-5f);
    const float ga = g[c], bb = be[c];
    float s = 0.f, mx = -3.4e38f;
    for (int sp = 0; sp < HO * WO; ++sp) {
      const float v = (x[(size_t)(bt * HO * WO + sp) * Cn + c] - mu) * inv * ga + bb;
      s += v; mx = fmaxf(mx, v);
    }
    pooled[(size_t)bt * C2 + c]      = s * (1.f / (HO * WO));
    pooled[(size_t)bt * C2 + Cn + c] = mx;
  }
}

// ---------------- Kernel 5: LayerNorm over last dim (C2) ----------------
__global__ void layernorm_k(const float* __restrict__ in, const float* __restrict__ g,
                            const float* __restrict__ be, float* __restrict__ out) {
  __shared__ float red[256];
  const float* row = in + (size_t)blockIdx.x * C2;
  float s = 0.f;
  for (int i = threadIdx.x; i < C2; i += 256) s += row[i];
  red[threadIdx.x] = s; __syncthreads();
  for (int o = 128; o > 0; o >>= 1) {
    if (threadIdx.x < o) red[threadIdx.x] += red[threadIdx.x + o];
    __syncthreads();
  }
  const float mu = red[0] / (float)C2;
  __syncthreads();
  float s2 = 0.f;
  for (int i = threadIdx.x; i < C2; i += 256) { const float d = row[i] - mu; s2 += d * d; }
  red[threadIdx.x] = s2; __syncthreads();
  for (int o = 128; o > 0; o >>= 1) {
    if (threadIdx.x < o) red[threadIdx.x] += red[threadIdx.x + o];
    __syncthreads();
  }
  const float inv = rsqrtf(red[0] / (float)C2 + 1e-5f);
  for (int i = threadIdx.x; i < C2; i += 256)
    out[(size_t)blockIdx.x * C2 + i] = (row[i] - mu) * inv * g[i] + be[i];
}

// ---------------- Kernel 6a/6b: global-half temporal mean, then concat ----------------
__global__ void gmean_k(const float* __restrict__ x1, float* __restrict__ gm) {
  const int b = blockIdx.x;
  const int j = threadIdx.x;       // blockDim = 384
  if (j < Cn) {
    float s = 0.f;
    for (int t = 0; t < Tn; ++t) s += x1[(size_t)(b * Tn + t) * C2 + Cn + j];
    gm[b * Cn + j] = s * (1.f / Tn);
  }
}

__global__ void xcat_k(const float* __restrict__ x1, const float* __restrict__ gm,
                       float* __restrict__ xc) {
  const int bt = blockIdx.x;
  const int b = bt / Tn;
  for (int j = threadIdx.x; j < C2; j += 256)
    xc[(size_t)bt * C2 + j] = (j < Cn) ? x1[(size_t)bt * C2 + j] : gm[b * Cn + (j - Cn)];
}

// ---------------- Kernel 7: score head (192 -> 1) + per-batch min-max normalize -------
__global__ void score_k(const float* __restrict__ h2, const float* __restrict__ w3,
                        const float* __restrict__ b3, float* __restrict__ score) {
  const int b = blockIdx.x;   // 32 blocks, 32 threads
  __shared__ float sc[Tn];
  const int t = threadIdx.x;
  if (t < Tn) {
    const float* row = h2 + (size_t)(b * Tn + t) * (Cn / 2);
    float s = b3[0];
    for (int j = 0; j < Cn / 2; ++j) s += row[j] * w3[j];
    sc[t] = s;
  }
  __syncthreads();
  if (t < Tn) {
    float mn = sc[0], mx = sc[0];
    for (int i = 1; i < Tn; ++i) { mn = fminf(mn, sc[i]); mx = fmaxf(mx, sc[i]); }
    score[b * Tn + t] = (sc[t] - mn) / (mx - mn + 1e-5f);
  }
}

// ---------------- Kernel 8: zero scratch ----------------
__global__ void zero_k(float* __restrict__ p, int n) {
  const int i = blockIdx.x * 256 + threadIdx.x;
  if (i < n) p[i] = 0.f;
}

// ---------------- Kernel 9: perturbed top-k / back-k indicator histograms -------------
__global__ void indic_k(const float* __restrict__ score, const float* __restrict__ noise,
                        float* __restrict__ ind_top, float* __restrict__ ind_back) {
  const int idx = blockIdx.x * 256 + threadIdx.x;
  if (idx >= Bn * NSAMP) return;
  const int b = idx / NSAMP, s = idx - b * NSAMP;
  float p[Tn];
#pragma unroll
  for (int t = 0; t < Tn; ++t)
    p[t] = score[b * Tn + t] + noise[(size_t)(b * NSAMP + s) * Tn + t] * SIGMA;
  int topmask = 0;
#pragma unroll
  for (int t = 0; t < Tn; ++t) {
    int r = 0;
#pragma unroll
    for (int j = 0; j < Tn; ++j)
      if (p[j] > p[t] || (p[j] == p[t] && j < t)) ++r;   // top_k tie-break: lower index
    if (r < KSEL) topmask |= (1 << t);
  }
  int kt = 0, kb = 0;
#pragma unroll
  for (int t = 0; t < Tn; ++t) {
    if (topmask & (1 << t)) { atomicAdd(&ind_top[(b * KSEL + kt) * Tn + t], 1e-3f);  ++kt; }
    else                    { atomicAdd(&ind_back[(b * KSEL + kb) * Tn + t], 1e-3f); ++kb; }
  }
}

// ---------------- Kernel 10: weighted gather over T (the einsum), b128 single pass ----
// grid (Cn*49/128, Bn), block 128. Each thread owns one (c, 4-wide hw group).
__global__ __launch_bounds__(128)
void gather_k(const float* __restrict__ frames, const float* __restrict__ ind_top,
              const float* __restrict__ ind_back, float* __restrict__ out_top,
              float* __restrict__ out_back) {
  __shared__ float wt[KSEL * Tn], wb[KSEL * Tn];
  const int b = blockIdx.y;
  wt[threadIdx.x] = ind_top[b * 128 + threadIdx.x];
  wb[threadIdx.x] = ind_back[b * 128 + threadIdx.x];
  __syncthreads();
  const int j = blockIdx.x * 128 + threadIdx.x;   // < Cn*49 = 18816
  const int c = j / 49, q = j - c * 49;
  const int hw = q * 4;
  float4 fv[Tn];
  const float* fp = frames + ((size_t)(b * Cn + c) * Tn) * (Hn * Wn) + hw;
#pragma unroll
  for (int t = 0; t < Tn; ++t) fv[t] = *(const float4*)&fp[t * (Hn * Wn)];
#pragma unroll
  for (int k = 0; k < KSEL; ++k) {
    float4 st = {0.f, 0.f, 0.f, 0.f}, sb = {0.f, 0.f, 0.f, 0.f};
#pragma unroll
    for (int t = 0; t < Tn; ++t) {
      const float a = wt[k * Tn + t], d = wb[k * Tn + t];
      st.x += a * fv[t].x; st.y += a * fv[t].y; st.z += a * fv[t].z; st.w += a * fv[t].w;
      sb.x += d * fv[t].x; sb.y += d * fv[t].y; sb.z += d * fv[t].z; sb.w += d * fv[t].w;
    }
    *(float4*)&out_top[((size_t)(b * Cn + c) * KSEL + k) * (Hn * Wn) + hw] = st;
    *(float4*)&out_back[((size_t)(b * Cn + c) * KSEL + k) * (Hn * Wn) + hw] = sb;
  }
}

// ---------------- Kernel 11: argsort(score) over T, ascending ----------------
__global__ void argsort_k(const float* __restrict__ score, int* __restrict__ out) {
  const int b = blockIdx.x, t = threadIdx.x;
  if (t < Tn) {
    const float v = score[b * Tn + t];
    int r = 0;
    for (int j = 0; j < Tn; ++j) {
      const float u = score[b * Tn + j];
      if (u < v || (u == v && j < t)) ++r;
    }
    out[b * Tn + r] = t;
  }
}

// ---------------- host launcher ----------------
extern "C" void kernel_launch(void* const* d_in, const int* in_sizes, int n_in,
                              void* d_out, int out_size, void* d_ws, size_t ws_size,
                              hipStream_t stream) {
  const float* frames = (const float*)d_in[0];
  const float* noise  = (const float*)d_in[1];
  const float* conv_w = (const float*)d_in[2];
  const float* conv_b = (const float*)d_in[3];
  const float* bn_g   = (const float*)d_in[4];
  const float* bn_b   = (const float*)d_in[5];
  const float* ln_g   = (const float*)d_in[6];
  const float* ln_b   = (const float*)d_in[7];
  const float* lin_w  = (const float*)d_in[8];
  const float* lin_b  = (const float*)d_in[9];
  const float* w1     = (const float*)d_in[10];
  const float* b1     = (const float*)d_in[11];
  const float* w2     = (const float*)d_in[12];
  const float* b2     = (const float*)d_in[13];
  const float* w3     = (const float*)d_in[14];
  const float* b3     = (const float*)d_in[15];

  float* ws = (float*)d_ws;
  float* conv_out = ws;                              // MCONV*Cn = 7077888
  float* bn_sum   = conv_out + (size_t)MCONV * Cn;   // 384
  float* bn_sum2  = bn_sum + Cn;                     // 384
  float* pooled   = bn_sum2 + Cn;                    // BT*C2
  float* xln      = pooled + (size_t)BT * C2;
  float* x1       = xln + (size_t)BT * C2;
  float* gmean    = x1 + (size_t)BT * C2;            // Bn*Cn
  float* xcat     = gmean + (size_t)Bn * Cn;
  float* h1       = xcat + (size_t)BT * C2;          // BT*Cn
  float* h2       = h1 + (size_t)BT * Cn;            // BT*Cn/2
  float* score    = h2 + (size_t)BT * (Cn / 2);      // BT
  float* ind_top  = score + BT;                      // Bn*KSEL*Tn
  float* ind_back = ind_top + Bn * KSEL * Tn;
  // pre-converted bf16 weights (2 bf16 per float slot; all 16B-aligned)
  float* wtail    = ind_back + Bn * KSEL * Tn;
  unsigned short* wg_conv = (unsigned short*)wtail;                  // KCONV*Cn
  unsigned short* wg_lin  = wg_conv + (size_t)KCONV * Cn;            // C2*C2
  unsigned short* wg_w1   = wg_lin + (size_t)C2 * C2;                // C2*Cn
  unsigned short* wg_w2   = wg_w1 + (size_t)C2 * Cn;                 // Cn*(Cn/2)

  float* out_top  = (float*)d_out;
  float* out_back = out_top + (size_t)FRAME_OUT;
  int*   out_idx  = (int*)(out_back + (size_t)FRAME_OUT);

  // 0. one-time weight conversion to bf16 (tiny; L2-resident thereafter)
  cvt_conv_w_k<<<(KCONV * Cn + 255) / 256, 256, 0, stream>>>(conv_w, wg_conv);
  cvt_w_k<<<(C2 * C2 + 255) / 256, 256, 0, stream>>>(lin_w, wg_lin, C2 * C2);
  cvt_w_k<<<(C2 * Cn + 255) / 256, 256, 0, stream>>>(w1, wg_w1, C2 * Cn);
  cvt_w_k<<<(Cn * (Cn / 2) + 255) / 256, 256, 0, stream>>>(w2, wg_w2, Cn * (Cn / 2));

  // 1. conv as implicit GEMM (WMMA bf16): M=18432, N=384, K=3456
  conv_wmma_k<<<dim3(Cn / BN, MCONV / BM), 256, 0, stream>>>(frames, wg_conv, conv_b,
                                                             conv_out);
  // 2. BN stats (zero accumulators, then coalesced partial sums)
  zero_k<<<(2 * Cn + 255) / 256, 256, 0, stream>>>(bn_sum, 2 * Cn);
  bn_stats_k<<<dim3(Cn / 64, 32), 256, 0, stream>>>(conv_out, bn_sum, bn_sum2);
  // 3. BN-normalize + spatial mean/max pool
  bnpool_k<<<BT, 256, 0, stream>>>(conv_out, bn_sum, bn_sum2, bn_g, bn_b, pooled);
  // 4. LayerNorm
  layernorm_k<<<BT, 256, 0, stream>>>(pooled, ln_g, ln_b, xln);
  // 5. lin (768x768) + GELU
  gemm_wmma_k<<<dim3(C2 / BN, BT / BM), 256, 0, stream>>>(xln, wg_lin, lin_b, x1,
                                                          BT, C2, C2, 1);
  // 6. local/global split
  gmean_k<<<Bn, 384, 0, stream>>>(x1, gmean);
  xcat_k<<<BT, 256, 0, stream>>>(x1, gmean, xcat);
  // 7. w1 (768->384) + GELU
  gemm_wmma_k<<<dim3(Cn / BN, BT / BM), 256, 0, stream>>>(xcat, wg_w1, b1, h1,
                                                          BT, Cn, C2, 1);
  // 8. w2 (384->192) + GELU
  gemm_wmma_k<<<dim3((Cn / 2) / BN, BT / BM), 256, 0, stream>>>(h1, wg_w2, b2, h2,
                                                                BT, Cn / 2, Cn, 1);
  // 9. score head + min-max normalize
  score_k<<<Bn, 32, 0, stream>>>(h2, w3, b3, score);
  // 10. indicators
  const int nind = 2 * Bn * KSEL * Tn;
  zero_k<<<(nind + 255) / 256, 256, 0, stream>>>(ind_top, nind);
  indic_k<<<(Bn * NSAMP + 255) / 256, 256, 0, stream>>>(score, noise, ind_top, ind_back);
  // 11. weighted gather (the einsum) — single b128 pass over frames
  gather_k<<<dim3((Cn * 49) / 128, Bn), 128, 0, stream>>>(frames, ind_top, ind_back,
                                                          out_top, out_back);
  // 12. argsort
  argsort_k<<<Bn, 32, 0, stream>>>(score, out_idx);
}
